// SinkhornDistance_49984829391475
// MI455X (gfx1250) — matrix operand order repaired and checked by reading, not compiled, not run on hardware
//
#include <hip/hip_runtime.h>
#include <hip/hip_bf16.h>
#include <math.h>

// ---------------------------------------------------------------------------
// Sinkhorn distance, MI455X (gfx1250, wave32, WMMA).
//   N=4 batches, P=1024 points, D=32 dims, eps=0.1, 30 iters, thresh=0.1.
// Cost matrix C (16 MB) is L2-resident (192 MB L2); built once with
// split-precision f16 WMMA (3x v_wmma_f32_16x16x32_f16 per 16x16 tile).
// Iterations: one-pass streaming logsumexp; all C accesses coalesced
// (row kernels: float4/lane; column kernel: lanes-across-columns + LDS merge).
// ---------------------------------------------------------------------------

typedef __attribute__((ext_vector_type(16))) _Float16 v16h;
typedef __attribute__((ext_vector_type(8)))  float    v8f;

#define NB 4
#define PP 1024
#define DD 32
#define EPSV     0.1f
#define INV_EPS  10.0f
#define THRESH   0.1f
#define MAX_ITER 30
#define NEG_BIG  (-3.0e38f)

__device__ __forceinline__ void lse_push(float& m, float& s, float val) {
    if (val <= m) { s += __expf(val - m); }
    else          { s = s * __expf(m - val) + 1.0f; m = val; }
}
__device__ __forceinline__ void lse_merge(float& m, float& s, float mo, float so) {
    const float M = fmaxf(m, mo);
    s = s * __expf(m - M) + so * __expf(mo - M);
    m = M;
}

// -------------------------------------------------------------------- init --
__global__ __launch_bounds__(1024)
void k_init(const float* __restrict__ xw, const float* __restrict__ yw,
            float* __restrict__ log_mu, float* __restrict__ log_nu,
            float* __restrict__ u, float* __restrict__ v, int* __restrict__ flag) {
    __shared__ float sx[1024];
    __shared__ float sy[1024];
    const int t = threadIdx.x;
    float a = 0.f, b = 0.f;
#pragma unroll
    for (int i = 0; i < 4; ++i) { a += xw[t + (i << 10)]; b += yw[t + (i << 10)]; }
    sx[t] = a; sy[t] = b;
    __syncthreads();
    for (int o = 512; o > 0; o >>= 1) {
        if (t < o) { sx[t] += sx[t + o]; sy[t] += sy[t + o]; }
        __syncthreads();
    }
    const float sumx = sx[0], sumy = sy[0];
#pragma unroll
    for (int i = 0; i < 4; ++i) {
        const int idx = t + (i << 10);
        log_mu[idx] = logf(xw[idx] / sumx + 1e-8f);
        log_nu[idx] = logf(yw[idx] / sumy + 1e-8f);
        u[idx] = 0.f;
        v[idx] = 0.f;
    }
    if (t == 0) *flag = 0;
}

// -------------------------------------------------------------------- gemm --
// C[n][p][q] = |x_p|^2 + |y_q|^2 - 2 * <x_p, y_q>.
// One wave per 16x16 tile; K=32 == D. Split-precision: hi*hi + hi*lo + lo*hi.
__global__ __launch_bounds__(256)
void k_gemm_cost(const float* __restrict__ x, const float* __restrict__ y,
                 float* __restrict__ C) {
    const int wave = (blockIdx.x << 3) + (threadIdx.x >> 5);
    const int l    = threadIdx.x & 31;
    const int tq   = wave & 63;
    const int tp   = (wave >> 6) & 63;
    const int n    = wave >> 12;
    const int half = l >> 4;        // lane half selects K sub-range 16h..16h+15 (this code's order)
    const int mr   = l & 15;        // A row / B column within tile
    const int p0   = tp << 4, q0 = tq << 4;

    const float* xrow = x + (((size_t)n * PP + p0 + mr) * DD);
    const float* yrow = y + (((size_t)n * PP + q0 + mr) * DD);
    const int ka = half << 4;       // A: K in {ka..ka+15}
    const int kb = half << 4;       // B: K in {kb..kb+15}

    float va[16], vb[16];
#pragma unroll
    for (int i = 0; i < 16; ++i) va[i] = xrow[ka + i];
#pragma unroll
    for (int i = 0; i < 16; ++i) vb[i] = yrow[kb + i];

    v16h ah, al, bh, bl;
    float ax2 = 0.f, by2 = 0.f;
#pragma unroll
    for (int i = 0; i < 16; ++i) {
        const float a = va[i], b = vb[i];
        ax2 += a * a;  by2 += b * b;
        _Float16 h = (_Float16)a; ah[i] = h; al[i] = (_Float16)(a - (float)h);
        h = (_Float16)b;          bh[i] = h; bl[i] = (_Float16)(b - (float)h);
    }
    // lanes l and l^16 hold complementary K halves of the same row/col
    const float x2f = ax2 + __shfl_xor(ax2, 16, 32);   // |x_(p0+mr)|^2
    const float y2f = by2 + __shfl_xor(by2, 16, 32);   // |y_(q0+mr)|^2

    v8f acc = {};
    acc = __builtin_amdgcn_wmma_f32_16x16x32_f16(false, ah, false, bh, (short)0, acc, false, false);
    acc = __builtin_amdgcn_wmma_f32_16x16x32_f16(false, ah, false, bl, (short)0, acc, false, false);
    acc = __builtin_amdgcn_wmma_f32_16x16x32_f16(false, al, false, bh, (short)0, acc, false, false);

#pragma unroll
    for (int r = 0; r < 8; ++r) {
        const int m    = r + (half << 3);                 // D layout: row per VGPR
        const float xn = __shfl(x2f, m, 32);              // gather row norm
        const float cv = xn + y2f - 2.0f * acc[r];
        C[(((size_t)n * PP + p0 + m) * PP) + q0 + mr] = cv;
    }
}

// ------------------------------------------------------------- u update ----
// One wave per row: streaming logsumexp, float4 per lane (global_load_b128).
__global__ __launch_bounds__(256)
void k_update_u(const float* __restrict__ C, const float* __restrict__ u,
                const float* __restrict__ v, const float* __restrict__ log_mu,
                float* __restrict__ u_new, float* __restrict__ du,
                const int* __restrict__ flag) {
    if (*flag) return;                                    // frozen (converged)
    const int row = (blockIdx.x << 3) + (threadIdx.x >> 5);   // n*P + p
    const int l   = threadIdx.x & 31;
    const int n   = row >> 10;
    const float* Crow = C + (size_t)row * PP;
    const float* vrow = v + ((size_t)n << 10);
    const float ui = u[row];

    float m = NEG_BIG, s = 0.f;
    for (int q = l << 2; q < PP; q += 128) {              // 16B/lane, 512B/wave per step, coalesced
        __builtin_prefetch(Crow + q + 1024, 0, 0);        // global_prefetch_b8
        const float4 c  = *(const float4*)(Crow + q);
        const float4 vv = *(const float4*)(vrow + q);
        lse_push(m, s, (ui + vv.x - c.x) * INV_EPS);
        lse_push(m, s, (ui + vv.y - c.y) * INV_EPS);
        lse_push(m, s, (ui + vv.z - c.z) * INV_EPS);
        lse_push(m, s, (ui + vv.w - c.w) * INV_EPS);
    }
#pragma unroll
    for (int off = 16; off >= 1; off >>= 1) {             // wave32 (m,s) combine
        const float mo = __shfl_xor(m, off, 32);
        const float so = __shfl_xor(s, off, 32);
        lse_merge(m, s, mo, so);
    }
    if (l == 0) {
        const float lse = m + logf(s);
        const float un  = EPSV * (log_mu[row] - lse) + ui;
        u_new[row] = un;
        du[row]    = fabsf(un - ui);
    }
}

// ------------------------------------------------------------- v update ----
// Lanes map to 32 adjacent columns (coalesced 128B row segments); the 8 waves
// of a block split the 1024 rows into 128-row chunks; LDS (m,s) merge.
__global__ __launch_bounds__(256)
void k_update_v(const float* __restrict__ C, const float* __restrict__ u_new,
                float* __restrict__ v, const float* __restrict__ log_nu,
                const int* __restrict__ flag) {
    if (*flag) return;
    __shared__ float sm_m[8][32];
    __shared__ float sm_s[8][32];
    const int blk  = blockIdx.x;                 // n*32 + qgroup
    const int n    = blk >> 5;
    const int q0   = (blk & 31) << 5;
    const int wave = threadIdx.x >> 5;
    const int l    = threadIdx.x & 31;
    const float* Cb   = C + (((size_t)n << 10) + (size_t)(wave << 7)) * PP + q0 + l;
    const float* urow = u_new + ((size_t)n << 10) + (wave << 7);
    const float vj = v[((size_t)n << 10) + q0 + l];

    float m = NEG_BIG, s = 0.f;
    for (int p = 0; p < 128; ++p) {
        __builtin_prefetch(Cb + (size_t)(p + 8) * PP, 0, 0);
        const float val = (urow[p] + vj - Cb[(size_t)p * PP]) * INV_EPS;
        lse_push(m, s, val);
    }
    sm_m[wave][l] = m;
    sm_s[wave][l] = s;
    __syncthreads();
    if (wave == 0) {
        float M = sm_m[0][l], S = sm_s[0][l];
#pragma unroll
        for (int w = 1; w < 8; ++w) lse_merge(M, S, sm_m[w][l], sm_s[w][l]);
        const float lse = M + logf(S);
        const int col = (n << 10) + q0 + l;
        v[col] = EPSV * (log_nu[col] - lse) + vj;
    }
}

// ------------------------------------------------------------ converge -----
__global__ __launch_bounds__(1024)
void k_check(const float* __restrict__ du, float* __restrict__ u,
             const float* __restrict__ u_new, int* __restrict__ flag) {
    __shared__ float sm[1024];
    const int t = threadIdx.x;
    float s = du[t] + du[t + 1024] + du[t + 2048] + du[t + 3072];
    sm[t] = s;
    __syncthreads();
    for (int o = 512; o > 0; o >>= 1) {
        if (t < o) sm[t] += sm[t + o];
        __syncthreads();
    }
    if (t == 0) {
        const float err = sm[0] * (1.0f / NB);
        if (err < THRESH) *flag = 1;
    }
#pragma unroll
    for (int i = 0; i < 4; ++i) {
        const int idx = t + (i << 10);
        u[idx] = u_new[idx];      // identity when frozen (u_new == u)
    }
}

// ---------------------------------------------------------------- cost -----
// rowsum[n*P+p] = sum_q exp((-C+u+v)/eps) * C, float4 per lane.
__global__ __launch_bounds__(256)
void k_cost(const float* __restrict__ C, const float* __restrict__ u,
            const float* __restrict__ v, float* __restrict__ rowsum) {
    const int row = (blockIdx.x << 3) + (threadIdx.x >> 5);
    const int l   = threadIdx.x & 31;
    const int n   = row >> 10;
    const float* Crow = C + (size_t)row * PP;
    const float* vrow = v + ((size_t)n << 10);
    const float ui = u[row];
    float s = 0.f;
    for (int q = l << 2; q < PP; q += 128) {
        __builtin_prefetch(Crow + q + 1024, 0, 0);
        const float4 c  = *(const float4*)(Crow + q);
        const float4 vv = *(const float4*)(vrow + q);
        s += __expf((ui + vv.x - c.x) * INV_EPS) * c.x;
        s += __expf((ui + vv.y - c.y) * INV_EPS) * c.y;
        s += __expf((ui + vv.z - c.z) * INV_EPS) * c.z;
        s += __expf((ui + vv.w - c.w) * INV_EPS) * c.w;
    }
#pragma unroll
    for (int off = 16; off >= 1; off >>= 1) s += __shfl_xor(s, off, 32);
    if (l == 0) rowsum[row] = s;
}

// Deterministic final reduction (no float atomics -> replay-stable bits).
__global__ __launch_bounds__(256)
void k_reduce(const float* __restrict__ rowsum, float* __restrict__ out) {
    __shared__ float sm[256];
    const int n = blockIdx.x, t = threadIdx.x;
    float s = 0.f;
    for (int i = t; i < PP; i += 256) s += rowsum[(size_t)n * PP + i];
    sm[t] = s;
    __syncthreads();
    for (int o = 128; o > 0; o >>= 1) {
        if (t < o) sm[t] += sm[t + o];
        __syncthreads();
    }
    if (t == 0) out[n] = sm[0];
}

// ---------------------------------------------------------------------------
extern "C" void kernel_launch(void* const* d_in, const int* in_sizes, int n_in,
                              void* d_out, int out_size, void* d_ws, size_t ws_size,
                              hipStream_t stream) {
    const float* x  = (const float*)d_in[0];   // (4,1024,32)
    const float* y  = (const float*)d_in[1];   // (4,1024,32)
    const float* xw = (const float*)d_in[2];   // (4,1024)
    const float* yw = (const float*)d_in[3];   // (4,1024)

    // Workspace layout (needs ~16.2 MB; C is L2-resident on MI455X's 192MB L2)
    float* C = (float*)d_ws;                       // 4*1024*1024 f32
    const size_t NC = (size_t)NB * PP * PP;
    const size_t NP = (size_t)NB * PP;
    float* u      = C + NC;
    float* u_new  = u + NP;
    float* v      = u_new + NP;
    float* log_mu = v + NP;
    float* log_nu = log_mu + NP;
    float* du     = log_nu + NP;
    float* rowsum = du + NP;
    int*   flag   = (int*)(rowsum + NP);

    k_init<<<1, 1024, 0, stream>>>(xw, yw, log_mu, log_nu, u, v, flag);

    // 16384 tiles / 8 waves per block = 2048 blocks
    k_gemm_cost<<<2048, 256, 0, stream>>>(x, y, C);

    for (int it = 0; it < MAX_ITER; ++it) {
        k_update_u<<<512, 256, 0, stream>>>(C, u, v, log_mu, u_new, du, flag);
        k_update_v<<<128, 256, 0, stream>>>(C, u_new, v, log_nu, flag);
        k_check<<<1, 1024, 0, stream>>>(du, u, u_new, flag);
    }

    k_cost<<<512, 256, 0, stream>>>(C, u, v, rowsum);
    k_reduce<<<NB, 256, 0, stream>>>(rowsum, (float*)d_out);
}
